// VAE_7765300871747
// MI455X (gfx1250) — matrix operand order repaired and checked by reading, not compiled, not run on hardware
//
#include <hip/hip_runtime.h>
#include <hip/hip_bf16.h>

// ---------------------------------------------------------------------------
// Latent-ODE VAE forward for MI455X (gfx1250, wave32, WMMA bf16->f32).
//
// Pipeline:
//   1) spline_kernel       : natural-cubic coefficients (Thomas solve)
//   2) pack_* kernels      : f32 weights -> bf16 WMMA B-fragments (once per call)
//   3) ode_kernel          : persistent per-16-row RK4 integration, all GEMMs
//                            via v_wmma_f32_16x16x32_bf16, f32 accumulate
//   4) readout_kernel      : elu(decode@W1+b1)@W2+b2 via WMMA
//
// Best-known configuration (round 2): hoisted loop-invariant A-fragments,
// union-bitcast fragment assembly, rcp/exp silu/tanh epilogues, SINGLE
// accumulator chain per tile (dual chains caused scratch spills).
// ---------------------------------------------------------------------------

typedef __bf16 bf16_t;
typedef __attribute__((ext_vector_type(8)))  float  v8f;
typedef __attribute__((ext_vector_type(8)))  bf16_t v8bf;
typedef __attribute__((ext_vector_type(16))) bf16_t v16bf;

#define B_SZ  1024
#define T_SZ  64
#define F_SZ  32
#define CIN   33
#define LDIM  64
#define TWOL  128
#define HDIM  256
#define OUTD  32
#define NINT  63          // T-1 intervals

__device__ inline bf16_t f2b(float f) {
  union { float f; unsigned u; } a; a.f = f;
  unsigned r = a.u + 0x7fffu + ((a.u >> 16) & 1u);   // RNE
  union { unsigned short s; bf16_t b; } o;
  o.s = (unsigned short)(r >> 16);
  return o.b;
}

// Fast epilogue math: v_exp_f32 + v_rcp_f32, no IEEE division sequences.
__device__ inline float fast_silu(float x) {
  return x * __builtin_amdgcn_rcpf(1.0f + __expf(-x));
}
__device__ inline float fast_tanh(float x) {
  // 1 - 2/(e^{2x}+1): correct saturation at +/-1 for large |x|
  return 1.0f - 2.0f * __builtin_amdgcn_rcpf(__expf(2.0f * x) + 1.0f);
}
__device__ inline float fast_elu(float x) {
  return (x > 0.f) ? x : (__expf(x) - 1.f);
}

// A-fragment for V_WMMA_*_16X16X32 (16-bit A, 16x32):
// lane M = lane&15; each lane holds K = base..base+15 (contiguous);
// base = (lane>>4)*16.  p points at A[row*strideK + kk*32 + base].
__device__ inline v16bf concat_frag(const bf16_t* p) {
  union { v8bf h[2]; v16bf v; } u;
  u.h[0] = *(const v8bf*)p;
  u.h[1] = *(const v8bf*)(p + 8);
  return u.v;
}
// Per-lane base pointer into a row-major bf16 A tile.
__device__ inline const bf16_t* a_lane_ptr(const bf16_t* A, int strideK, int lane) {
  return A + (lane & 15) * strideK + (lane >> 4) * 16;
}

// One MLP layer tile-GEMM: (16 x K) @ (K x N) + bias, silu, bf16 out.
// Wave w owns N-tiles w, w+8, ... ; A-fragments hoisted across N-tiles.
template <int K>
__device__ inline void gemm_layer(const bf16_t* A, const bf16_t* Wp,
                                  const float* bias, int N, bf16_t* Aout,
                                  int lane, int wave) {
  const int nnt = N >> 4;
  constexpr int nkt = K >> 5;
  const bf16_t* ap = a_lane_ptr(A, K, lane);
  v16bf af[nkt];
#pragma unroll
  for (int kk = 0; kk < nkt; ++kk) af[kk] = concat_frag(ap + kk * 32);

  for (int nt = wave; nt < nnt; nt += 8) {
    v8f acc = {};
#pragma unroll
    for (int kk = 0; kk < nkt; ++kk) {
      v16bf b = *(const v16bf*)(Wp + ((size_t)(kk * nnt + nt)) * 512 + lane * 16);
      acc = __builtin_amdgcn_wmma_f32_16x16x32_bf16(false, af[kk], false, b,
                                                    (short)0, acc, false, false);
    }
    const int col   = nt * 16 + (lane & 15);
    const int rbase = (lane >> 4) * 8;
    const float bv  = bias[col];
#pragma unroll
    for (int e2 = 0; e2 < 8; ++e2)
      Aout[(rbase + e2) * N + col] = f2b(fast_silu(acc[e2] + bv));
  }
}

// ---------------------------------------------------------------------------
// 1) Natural cubic spline coefficients.  One thread per (b, ch).
//    coef layout: [interval t][b*33+ch]  (coalesced for the ODE kernel).
// ---------------------------------------------------------------------------
__global__ __launch_bounds__(256) void spline_kernel(const float* __restrict__ data,
                                                     float* __restrict__ cbo,
                                                     float* __restrict__ cco,
                                                     float* __restrict__ cdo) {
  const int id = blockIdx.x * blockDim.x + threadIdx.x;
  if (id >= B_SZ * CIN) return;
  const int b = id / CIN, ch = id % CIN;

  float yv[T_SZ];
  if (ch == 0) {
    for (int t = 0; t < T_SZ; ++t) yv[t] = (float)t * (1.0f / T_SZ);
  } else {
    const float* p = data + (size_t)b * T_SZ * F_SZ + (ch - 1);
    for (int t = 0; t < T_SZ; ++t) yv[t] = p[t * F_SZ];
  }
  // Thomas solve of (4,1) tridiagonal, 62 interior nodes
  float cp[62], dp[62];
  {
    float r0 = 6.f * (yv[0] - 2.f * yv[1] + yv[2]);
    cp[0] = 0.25f; dp[0] = r0 * 0.25f;
    for (int i = 1; i < 62; ++i) {
      float ri = 6.f * (yv[i] - 2.f * yv[i + 1] + yv[i + 2]);
      float m = 4.f - cp[i - 1];
      cp[i] = 1.f / m;
      dp[i] = (ri - dp[i - 1]) * cp[i];
    }
  }
  float M[T_SZ];
  M[0] = 0.f; M[T_SZ - 1] = 0.f;
  M[62] = dp[61];
  for (int i = 60; i >= 0; --i) M[i + 1] = dp[i] - cp[i] * M[i + 2];

  for (int i = 0; i < NINT; ++i) {
    float bc = yv[i + 1] - yv[i] - (2.f * M[i] + M[i + 1]) * (1.f / 6.f);
    float cc = M[i] * 0.5f;
    float dc = (M[i + 1] - M[i]) * (1.f / 6.f);
    size_t o = (size_t)i * (B_SZ * CIN) + id;
    cbo[o] = bc; cco[o] = cc; cdo[o] = dc;
  }
}

// ---------------------------------------------------------------------------
// 2) Weight packing into WMMA B-fragment order.
//    Fragment f=(kt*nnt+nt): 512 bf16; lane owns 16 consecutive-K of its column:
//    packed[f*512 + lane*16 + j] = W[kt*32 + (lane>>4)*16 + j][nt*16 + (lane&15)]
// ---------------------------------------------------------------------------
__global__ void pack_w_kernel(const float* __restrict__ W, bf16_t* __restrict__ dst,
                              int K, int N) {
  const int f = blockIdx.x * 8 + (threadIdx.x >> 5);
  const int lane = threadIdx.x & 31;
  const int nnt = N >> 4, nkt = K >> 5;
  if (f >= nkt * nnt) return;
  const int kt = f / nnt, nt = f % nnt;
  const int col = nt * 16 + (lane & 15);
  const int kb = kt * 32 + (lane >> 4) * 16;
  bf16_t* o = dst + (size_t)f * 512 + lane * 16;
#pragma unroll
  for (int j = 0; j < 16; ++j) o[j] = f2b(W[(size_t)(kb + j) * N + col]);
}

// Encoder head: 33 stride-column slices of concat[Wm|Ws] (K=256, N=128 each).
// Slice c column h <- (h<64 ? Wm[:,h*33+c] : Ws[:,(h-64)*33+c]).
__global__ void pack_enc_head(const float* __restrict__ Wm, const float* __restrict__ Ws,
                              const float* __restrict__ bm, const float* __restrict__ bs,
                              bf16_t* __restrict__ dst, float* __restrict__ bdst) {
  const int f = blockIdx.x * 8 + (threadIdx.x >> 5);
  const int lane = threadIdx.x & 31;
  if (f >= 33 * 64) return;
  const int c = f / 64, kt = (f % 64) / 8, nt = f % 8;
  const int h = nt * 16 + (lane & 15);
  const float* W = (h < 64) ? Wm : Ws;
  const int srccol = (h < 64) ? (h * 33 + c) : ((h - 64) * 33 + c);
  const int kb = kt * 32 + (lane >> 4) * 16;
  bf16_t* o = dst + (size_t)f * 512 + lane * 16;
#pragma unroll
  for (int j = 0; j < 16; ++j) o[j] = f2b(W[(size_t)(kb + j) * 2112 + srccol]);
  if (kt == 0 && lane < 16)
    bdst[c * 128 + h] = (h < 64) ? bm[h * 33 + c] : bs[(h - 64) * 33 + c];
}

// Decoder head: 65 slices of Wf (K=256, N=32 each); column o <- Wf[:, o*65+c].
__global__ void pack_dec_head(const float* __restrict__ Wf, const float* __restrict__ bfv,
                              bf16_t* __restrict__ dst, float* __restrict__ bdst) {
  const int f = blockIdx.x * 8 + (threadIdx.x >> 5);
  const int lane = threadIdx.x & 31;
  if (f >= 65 * 16) return;
  const int c = f / 16, kt = (f % 16) / 2, nt = f % 2;
  const int oo = nt * 16 + (lane & 15);
  const int srccol = oo * 65 + c;
  const int kb = kt * 32 + (lane >> 4) * 16;
  bf16_t* o = dst + (size_t)f * 512 + lane * 16;
#pragma unroll
  for (int j = 0; j < 16; ++j) o[j] = f2b(Wf[(size_t)(kb + j) * 2080 + srccol]);
  if (kt == 0 && lane < 16) bdst[c * 32 + oo] = bfv[srccol];
}

// ---------------------------------------------------------------------------
// 3) Persistent RK4 ODE kernel. 16 batch rows / workgroup, 8 waves.
// ---------------------------------------------------------------------------
__global__ __launch_bounds__(256) void ode_kernel(
    const float* __restrict__ data,
    const float* __restrict__ cbo, const float* __restrict__ cco, const float* __restrict__ cdo,
    const float* __restrict__ iew, const float* __restrict__ ieb,
    const float* __restrict__ idw, const float* __restrict__ idb,
    const bf16_t* __restrict__ pE1, const float* __restrict__ be1,
    const bf16_t* __restrict__ pE2, const float* __restrict__ be2,
    const bf16_t* __restrict__ pE3, const float* __restrict__ be3,
    const bf16_t* __restrict__ pEH, const float* __restrict__ bEH,
    const bf16_t* __restrict__ pD1, const float* __restrict__ bd1,
    const bf16_t* __restrict__ pD2, const float* __restrict__ bd2,
    const bf16_t* __restrict__ pD3, const float* __restrict__ bd3,
    const bf16_t* __restrict__ pDH, const float* __restrict__ bDH,
    float* __restrict__ decode, float* __restrict__ klOut) {
  __shared__ float  sE[16][TWOL];     // state z_enc
  __shared__ float  sD[16][OUTD];     // state z_dec
  __shared__ float  sK[16];           // state kl
  __shared__ float  uE[16][TWOL];     // RK4 weighted k-sum
  __shared__ float  uD[16][OUTD];
  __shared__ float  uK[16];
  __shared__ bf16_t eEb[16][TWOL];    // eval point (A operand, enc L1)
  __shared__ bf16_t eDb[16][OUTD];    // eval point (A operand, dec L1)
  __shared__ bf16_t hA[16][HDIM];     // activation ping
  __shared__ bf16_t hB[16][HDIM];     // activation pong
  __shared__ float  kE[16][TWOL];     // k_cur z_enc
  __shared__ float  kD[16][OUTD];     // k_cur z_dec (atomic-reduced)
  __shared__ float  dXt[CIN][16];     // dX transposed [c][row]
  __shared__ float  dYt[LDIM + 1][16];// dY transposed [c][row]

  const int tid  = threadIdx.x;
  const int lane = tid & 31;
  const int wave = tid >> 5;
  const int row0 = blockIdx.x * 16;

  // ---- init: z_enc0 = X0@iew+ieb, z_dec0 = X0@idw+idb (X0[0]=0, X0[1:]=data[:,0,:])
  for (int idx = tid; idx < 16 * TWOL; idx += 256) {
    int r = idx >> 7, h = idx & 127;
    float s = ieb[h];
    const float* xr = data + (size_t)(row0 + r) * (T_SZ * F_SZ);
    for (int c = 1; c < CIN; ++c) s += xr[c - 1] * iew[c * TWOL + h];
    sE[r][h] = s; eEb[r][h] = f2b(s); uE[r][h] = 0.f;
  }
  for (int idx = tid; idx < 16 * OUTD; idx += 256) {
    int r = idx >> 5, o = idx & 31;
    float s = idb[o];
    const float* xr = data + (size_t)(row0 + r) * (T_SZ * F_SZ);
    for (int c = 1; c < CIN; ++c) s += xr[c - 1] * idw[c * OUTD + o];
    sD[r][o] = s; eDb[r][o] = f2b(s); uD[r][o] = 0.f;
    decode[(size_t)(row0 + r) * (T_SZ * OUTD) + o] = s;
  }
  if (tid < 16) { sK[tid] = 0.f; uK[tid] = 0.f; }
  __syncthreads();

  for (int t = 0; t < NINT; ++t) {
    for (int s = 0; s < 4; ++s) {
      const float u    = (s == 0) ? 0.f : ((s == 3) ? 1.f : 0.5f);
      const float cws  = (s == 1 || s == 2) ? 2.f : 1.f;       // RK4 sum weight
      const float wnxt = (s == 2) ? 1.f : 0.5f;                // eval weight for s+1

      // ---- dX (and dY[0]) for this stage; zero kD accumulator
      for (int idx = tid; idx < 16 * CIN; idx += 256) {
        int r = idx / CIN, c = idx % CIN;
        size_t o = (size_t)t * (B_SZ * CIN) + (size_t)(row0 + r) * CIN + c;
        float v = cbo[o] + u * (2.f * cco[o] + 3.f * cdo[o] * u);
        dXt[c][r] = v;
        if (c == 0) dYt[0][r] = v;
      }
      for (int idx = tid; idx < 16 * OUTD; idx += 256)
        kD[idx >> 5][idx & 31] = 0.f;
      // ---- dkl at the eval point
      if (tid < 16) {
        float acc = 0.f;
        for (int l = 0; l < LDIM; ++l) {
          float mu = (float)eEb[tid][l];
          float ls = fminf(4.f, fmaxf(-4.f, (float)eEb[tid][LDIM + l]));
          acc += mu * mu + __expf(2.f * ls) - 1.f - 2.f * ls;
        }
        uK[tid] += cws * (0.5f * acc * (1.f / LDIM));
      }
      __syncthreads();

      // ---- encoder MLP
      gemm_layer<TWOL>(&eEb[0][0], pE1, be1, HDIM, &hA[0][0], lane, wave);
      __syncthreads();
      gemm_layer<HDIM>(&hA[0][0], pE2, be2, HDIM, &hB[0][0], lane, wave);
      __syncthreads();
      gemm_layer<HDIM>(&hB[0][0], pE3, be3, HDIM, &hA[0][0], lane, wave);
      __syncthreads();

      // ---- encoder head: wave owns h-tile 'wave'; loop over 33 c-slices,
      //      A-fragments hoisted; tanh + dX contraction folded in registers.
      {
        const bf16_t* ap = a_lane_ptr(&hA[0][0], HDIM, lane);
        v16bf af[8];
#pragma unroll
        for (int kk = 0; kk < 8; ++kk) af[kk] = concat_frag(ap + kk * 32);

        v8f dze = {};
        const int col = wave * 16 + (lane & 15);
        const int rbase = (lane >> 4) * 8;
        for (int c = 0; c < CIN; ++c) {
          if (c + 1 < CIN)
            __builtin_prefetch(pEH + ((size_t)(((c + 1) * 8) * 8 + wave)) * 512 + lane * 16, 0, 1);
          v8f g = {};
#pragma unroll
          for (int kk = 0; kk < 8; ++kk) {
            v16bf b = *(const v16bf*)(pEH + ((size_t)((c * 8 + kk) * 8 + wave)) * 512 + lane * 16);
            g = __builtin_amdgcn_wmma_f32_16x16x32_bf16(false, af[kk], false, b,
                                                        (short)0, g, false, false);
          }
          const float bv = bEH[c * TWOL + col];
          const float* dxp = &dXt[c][rbase];
#pragma unroll
          for (int e2 = 0; e2 < 8; ++e2)
            dze[e2] += fast_tanh(g[e2] + bv) * dxp[e2];
        }
#pragma unroll
        for (int e2 = 0; e2 < 8; ++e2) {
          kE[rbase + e2][col] = dze[e2];
          if (wave < 4) dYt[1 + col][rbase + e2] = dze[e2];   // dY[1+l] = dze[l], l<64
        }
      }
      __syncthreads();

      // ---- decoder MLP
      gemm_layer<OUTD>(&eDb[0][0], pD1, bd1, HDIM, &hB[0][0], lane, wave);
      __syncthreads();
      gemm_layer<HDIM>(&hB[0][0], pD2, bd2, HDIM, &hA[0][0], lane, wave);
      __syncthreads();
      gemm_layer<HDIM>(&hA[0][0], pD3, bd3, HDIM, &hB[0][0], lane, wave);
      __syncthreads();

      // ---- decoder head: 65 c-slices x 2 N-tiles; wave -> (ntile, slice set)
      {
        const bf16_t* ap = a_lane_ptr(&hB[0][0], HDIM, lane);
        v16bf af[8];
#pragma unroll
        for (int kk = 0; kk < 8; ++kk) af[kk] = concat_frag(ap + kk * 32);

        const int nt = wave & 1;
        const int g0 = wave >> 1;
        const int col = nt * 16 + (lane & 15);
        const int rbase = (lane >> 4) * 8;
        v8f dzd = {};
        for (int c = g0; c <= LDIM; c += 4) {
          v8f q = {};
#pragma unroll
          for (int kk = 0; kk < 8; ++kk) {
            v16bf b = *(const v16bf*)(pDH + ((size_t)((c * 8 + kk) * 2 + nt)) * 512 + lane * 16);
            q = __builtin_amdgcn_wmma_f32_16x16x32_bf16(false, af[kk], false, b,
                                                        (short)0, q, false, false);
          }
          const float bv = bDH[c * OUTD + col];
          const float* dyp = &dYt[c][rbase];
#pragma unroll
          for (int e2 = 0; e2 < 8; ++e2)
            dzd[e2] += fast_tanh(q[e2] + bv) * dyp[e2];
        }
#pragma unroll
        for (int e2 = 0; e2 < 8; ++e2)
          atomicAdd(&kD[rbase + e2][col], dzd[e2]);
      }
      __syncthreads();

      // ---- accumulate RK4 sum, build next eval point
      for (int idx = tid; idx < 16 * TWOL; idx += 256) {
        int r = idx >> 7, h = idx & 127;
        float k = kE[r][h];
        uE[r][h] += cws * k;
        if (s < 3) eEb[r][h] = f2b(sE[r][h] + wnxt * k);
      }
      for (int idx = tid; idx < 16 * OUTD; idx += 256) {
        int r = idx >> 5, o = idx & 31;
        float k = kD[r][o];
        uD[r][o] += cws * k;
        if (s < 3) eDb[r][o] = f2b(sD[r][o] + wnxt * k);
      }
      __syncthreads();
    } // stages

    // ---- advance state, emit decode[:, t+1, :], rebuild stage-0 eval
    for (int idx = tid; idx < 16 * TWOL; idx += 256) {
      int r = idx >> 7, h = idx & 127;
      float ns = sE[r][h] + uE[r][h] * (1.f / 6.f);
      sE[r][h] = ns; eEb[r][h] = f2b(ns); uE[r][h] = 0.f;
    }
    for (int idx = tid; idx < 16 * OUTD; idx += 256) {
      int r = idx >> 5, o = idx & 31;
      float ns = sD[r][o] + uD[r][o] * (1.f / 6.f);
      sD[r][o] = ns; eDb[r][o] = f2b(ns); uD[r][o] = 0.f;
      decode[(size_t)(row0 + r) * (T_SZ * OUTD) + (size_t)(t + 1) * OUTD + o] = ns;
    }
    if (tid < 16) { sK[tid] += uK[tid] * (1.f / 6.f); uK[tid] = 0.f; }
    __syncthreads();
  } // intervals

  if (tid < 16) klOut[row0 + tid] = sK[tid];
}

// ---------------------------------------------------------------------------
// 4) Readout: pred = elu(decode @ W1 + b1) @ W2 + b2 over 65536 rows.
// ---------------------------------------------------------------------------
__global__ __launch_bounds__(256) void readout_kernel(
    const float* __restrict__ decode,
    const bf16_t* __restrict__ pW1, const float* __restrict__ rb1,
    const bf16_t* __restrict__ pW2, const float* __restrict__ rb2,
    float* __restrict__ out) {
  __shared__ bf16_t aIn[16][32];
  __shared__ bf16_t h1[16][TWOL];
  const int tid = threadIdx.x, lane = tid & 31, wave = tid >> 5;
  const size_t gr0 = (size_t)blockIdx.x * 16;

  for (int idx = tid; idx < 16 * 32; idx += 256) {
    int r = idx >> 5, c = idx & 31;
    aIn[r][c] = f2b(decode[(gr0 + r) * 32 + c]);
  }
  __syncthreads();
  {  // L1: (16x32)@(32x128) + b1, elu
    v8f acc = {};
    v16bf a = concat_frag(a_lane_ptr(&aIn[0][0], 32, lane));
    v16bf b = *(const v16bf*)(pW1 + (size_t)wave * 512 + lane * 16);
    acc = __builtin_amdgcn_wmma_f32_16x16x32_bf16(false, a, false, b,
                                                  (short)0, acc, false, false);
    const int col = wave * 16 + (lane & 15);
    const int rbase = (lane >> 4) * 8;
    const float bv = rb1[col];
#pragma unroll
    for (int e2 = 0; e2 < 8; ++e2)
      h1[rbase + e2][col] = f2b(fast_elu(acc[e2] + bv));
  }
  __syncthreads();
  if (wave < 2) {  // L2: (16x128)@(128x32) + b2
    const bf16_t* ap = a_lane_ptr(&h1[0][0], TWOL, lane);
    v8f acc = {};
#pragma unroll
    for (int kk = 0; kk < 4; ++kk) {
      v16bf a = concat_frag(ap + kk * 32);
      v16bf b = *(const v16bf*)(pW2 + (size_t)(kk * 2 + wave) * 512 + lane * 16);
      acc = __builtin_amdgcn_wmma_f32_16x16x32_bf16(false, a, false, b,
                                                    (short)0, acc, false, false);
    }
    const int col = wave * 16 + (lane & 15);
    const int rbase = (lane >> 4) * 8;
    const float bv = rb2[col];
#pragma unroll
    for (int e2 = 0; e2 < 8; ++e2)
      out[(gr0 + rbase + e2) * 32 + col] = acc[e2] + bv;
  }
}

// ---------------------------------------------------------------------------
// Host side
// ---------------------------------------------------------------------------
extern "C" void kernel_launch(void* const* d_in, const int* in_sizes, int n_in,
                              void* d_out, int out_size, void* d_ws, size_t ws_size,
                              hipStream_t stream) {
  (void)in_sizes; (void)n_in; (void)out_size; (void)ws_size;
  // setup_inputs() order (dicts in insertion order, depth-first):
  const float* data = (const float*)d_in[0];
  const float* iew  = (const float*)d_in[1];   // (33,128)
  const float* ieb  = (const float*)d_in[2];
  const float* idw  = (const float*)d_in[3];   // (33,32)
  const float* idb  = (const float*)d_in[4];
  const float* eW1  = (const float*)d_in[5];   // (128,256)
  const float* eb1  = (const float*)d_in[6];
  const float* eW2  = (const float*)d_in[7];   // (256,256)
  const float* eb2  = (const float*)d_in[8];
  const float* eW3  = (const float*)d_in[9];
  const float* eb3  = (const float*)d_in[10];
  const float* Wm   = (const float*)d_in[11];  // (256,2112)
  const float* bm   = (const float*)d_in[12];
  const float* Ws   = (const float*)d_in[13];
  const float* bs   = (const float*)d_in[14];
  const float* dW1  = (const float*)d_in[15];  // (32,256)
  const float* db1  = (const float*)d_in[16];
  const float* dW2  = (const float*)d_in[17];
  const float* db2  = (const float*)d_in[18];
  const float* dW3  = (const float*)d_in[19];
  const float* db3  = (const float*)d_in[20];
  const float* Wf   = (const float*)d_in[21];  // (256,2080)
  const float* bfv  = (const float*)d_in[22];
  const float* rW1  = (const float*)d_in[23];  // (32,128)
  const float* rb1  = (const float*)d_in[24];
  const float* rW2  = (const float*)d_in[25];  // (128,32)
  const float* rb2  = (const float*)d_in[26];

  char* ws = (char*)d_ws;
  size_t ofs = 0;
  auto give = [&](size_t bytes) -> char* {
    char* p = ws + ofs;
    ofs += (bytes + 255) & ~(size_t)255;
    return p;
  };
  const size_t COEF = (size_t)NINT * B_SZ * CIN;
  float* cb     = (float*)give(COEF * 4);
  float* cc     = (float*)give(COEF * 4);
  float* cd     = (float*)give(COEF * 4);
  float* decode = (float*)give((size_t)B_SZ * T_SZ * OUTD * 4);
  bf16_t* pE1 = (bf16_t*)give((size_t)4  * 16 * 512 * 2);
  bf16_t* pE2 = (bf16_t*)give((size_t)8  * 16 * 512 * 2);
  bf16_t* pE3 = (bf16_t*)give((size_t)8  * 16 * 512 * 2);
  bf16_t* pEH = (bf16_t*)give((size_t)33 * 64 * 512 * 2);
  bf16_t* pD1 = (bf16_t*)give((size_t)1  * 16 * 512 * 2);
  bf16_t* pD2 = (bf16_t*)give((size_t)8  * 16 * 512 * 2);
  bf16_t* pD3 = (bf16_t*)give((size_t)8  * 16 * 512 * 2);
  bf16_t* pDH = (bf16_t*)give((size_t)65 * 16 * 512 * 2);
  bf16_t* pR1 = (bf16_t*)give((size_t)8  * 512 * 2);
  bf16_t* pR2 = (bf16_t*)give((size_t)8  * 512 * 2);
  float*  bEH = (float*)give((size_t)33 * 128 * 4);
  float*  bDH = (float*)give((size_t)65 * 32 * 4);

  // 1) spline coefficients
  spline_kernel<<<(B_SZ * CIN + 255) / 256, 256, 0, stream>>>(data, cb, cc, cd);

  // 2) weight packing
  pack_w_kernel<<<8,   256, 0, stream>>>(eW1, pE1, 128, 256);
  pack_w_kernel<<<16,  256, 0, stream>>>(eW2, pE2, 256, 256);
  pack_w_kernel<<<16,  256, 0, stream>>>(eW3, pE3, 256, 256);
  pack_w_kernel<<<2,   256, 0, stream>>>(dW1, pD1, 32, 256);
  pack_w_kernel<<<16,  256, 0, stream>>>(dW2, pD2, 256, 256);
  pack_w_kernel<<<16,  256, 0, stream>>>(dW3, pD3, 256, 256);
  pack_w_kernel<<<1,   256, 0, stream>>>(rW1, pR1, 32, 128);
  pack_w_kernel<<<1,   256, 0, stream>>>(rW2, pR2, 128, 32);
  pack_enc_head<<<264, 256, 0, stream>>>(Wm, Ws, bm, bs, pEH, bEH);
  pack_dec_head<<<130, 256, 0, stream>>>(Wf, bfv, pDH, bDH);

  // 3) persistent RK4 integration (64 workgroups x 16 rows)
  float* predOut = (float*)d_out;
  float* klOut   = predOut + (size_t)B_SZ * T_SZ * OUTD;
  ode_kernel<<<B_SZ / 16, 256, 0, stream>>>(
      data, cb, cc, cd, iew, ieb, idw, idb,
      pE1, eb1, pE2, eb2, pE3, eb3, pEH, bEH,
      pD1, db1, pD2, db2, pD3, db3, pDH, bDH,
      decode, klOut);

  // 4) readout over B*T rows
  readout_kernel<<<(B_SZ * T_SZ) / 16, 256, 0, stream>>>(decode, pR1, rb1, pR2, rb2, predOut);
}
